// DualCoupledBehaviorAttention_52785148068652
// MI455X (gfx1250) — matrix operand order repaired and checked by reading, not compiled, not run on hardware
//
#include <hip/hip_runtime.h>
#include <hip/hip_bf16.h>

#define DD 256
#define LL 512
#define NB 64
#define SCALE 0.0625f
#define EPS 1e-8f

typedef __attribute__((ext_vector_type(16))) __bf16 v16bf;
typedef __attribute__((ext_vector_type(8)))  __bf16 v8bf;
typedef __attribute__((ext_vector_type(8)))  float  v8f;

// ---------- WMMA helpers (CDNA5 wave32, 16x16x32 bf16 -> f32) ----------

__device__ inline v8f wmma_bf16(v16bf a, v16bf b, v8f c) {
    return __builtin_amdgcn_wmma_f32_16x16x32_bf16(
        /*neg_a=*/false, a, /*neg_b=*/false, b,
        /*c_mod=*/(short)0, c, /*reuse_a=*/false, /*reuse_b=*/false);
}

// A-matrix fragment 16x32 (MxK), row-major source with leading dim ld.
// ISA layout: lane m=lane%16; lanes 0-15 hold K=[0..7]+[16..23], lanes 16-31 K=[8..15]+[24..31].
__device__ inline v16bf load_a(const __bf16* base, int ld, int m0, int k0) {
    int lane = threadIdx.x & 31;
    int m  = lane & 15;
    int kb = (lane & 16) ? 8 : 0;
    const __bf16* p = base + (size_t)(m0 + m) * ld + k0 + kb;
    v8bf lo = *(const v8bf*)(p);
    v8bf hi = *(const v8bf*)(p + 16);
    v16bf a;
#pragma unroll
    for (int i = 0; i < 8; ++i) { a[i] = lo[i]; a[i + 8] = hi[i]; }
    return a;
}

// B-matrix fragment 32x16 (KxN) taken from a transposed store Bt[n][k] (leading dim ld).
// ISA layout: column n=lane%16; lanes 0-15 hold K=[0..7]+[16..23], lanes 16-31 K=[8..15]+[24..31].
__device__ inline v16bf load_bt(const __bf16* baseT, int ld, int n0, int k0) {
    int lane = threadIdx.x & 31;
    int n  = lane & 15;
    int kb = (lane & 16) ? 8 : 0;
    const __bf16* p = baseT + (size_t)(n0 + n) * ld + k0 + kb;
    v8bf lo = *(const v8bf*)(p);
    v8bf hi = *(const v8bf*)(p + 16);
    v16bf b;
#pragma unroll
    for (int i = 0; i < 8; ++i) { b[i] = lo[i]; b[i + 8] = hi[i]; }
    return b;
}

__device__ inline float wave_sum(float v) {
#pragma unroll
    for (int m = 16; m >= 1; m >>= 1) v += __shfl_xor(v, m, 32);
    return v;
}
__device__ inline float wave_max(float v) {
#pragma unroll
    for (int m = 16; m >= 1; m >>= 1) v = fmaxf(v, __shfl_xor(v, m, 32));
    return v;
}

// ---------- stage 1: elementwise + LN + bf16 cast + code packing ----------

__global__ __launch_bounds__(256)
void prep_kernel(const float* __restrict__ item, const float* __restrict__ pos,
                 const float* __restrict__ bse, const int* __restrict__ bs_seq,
                 const float* __restrict__ lnw_ib, const float* __restrict__ lnb_ib,
                 const float* __restrict__ lnw_pb, const float* __restrict__ lnb_pb,
                 __bf16* __restrict__ Xib, __bf16* __restrict__ Xpb,
                 __bf16* __restrict__ LNib, __bf16* __restrict__ LNpb,
                 unsigned char* __restrict__ codes)
{
    int row = blockIdx.x;                  // 0..B*L-1
    int d   = threadIdx.x;                 // 0..255
    size_t idx = (size_t)row * DD + d;
    float xi = item[idx] + bse[idx];
    float xp = pos[idx]  + bse[idx];
    Xib[idx] = (__bf16)xi;
    Xpb[idx] = (__bf16)xp;

    __shared__ float red[2][8];
    int wid = threadIdx.x >> 5, lane = threadIdx.x & 31;

    float si = wave_sum(xi), sp = wave_sum(xp);
    if (lane == 0) { red[0][wid] = si; red[1][wid] = sp; }
    __syncthreads();
    float mi = 0.f, mp = 0.f;
#pragma unroll
    for (int i = 0; i < 8; ++i) { mi += red[0][i]; mp += red[1][i]; }
    mi *= (1.0f / DD); mp *= (1.0f / DD);
    __syncthreads();

    float di = xi - mi, dp = xp - mp;
    float vi = wave_sum(di * di), vp = wave_sum(dp * dp);
    if (lane == 0) { red[0][wid] = vi; red[1][wid] = vp; }
    __syncthreads();
    float svi = 0.f, svp = 0.f;
#pragma unroll
    for (int i = 0; i < 8; ++i) { svi += red[0][i]; svp += red[1][i]; }
    float ri = rsqrtf(svi * (1.0f / DD) + EPS);
    float rp = rsqrtf(svp * (1.0f / DD) + EPS);
    LNib[idx] = (__bf16)(lnw_ib[d] * di * ri + lnb_ib[d]);
    LNpb[idx] = (__bf16)(lnw_pb[d] * dp * rp + lnb_pb[d]);

    if (d == 0) {
        const int* q = bs_seq + (size_t)row * 4;
        codes[row] = (unsigned char)((q[0] & 1) | ((q[1] & 1) << 1) |
                                     ((q[2] & 1) << 2) | ((q[3] & 1) << 3));
    }
}

__global__ void transpose_w_kernel(const float* W0, const float* W1, const float* W2,
                                   const float* W3, const float* W4, const float* W5,
                                   const float* W6, __bf16* __restrict__ Wt)
{
    int w = blockIdx.y, n = blockIdx.x, k = threadIdx.x;
    const float* W;
    switch (w) {
        case 0: W = W0; break; case 1: W = W1; break; case 2: W = W2; break;
        case 3: W = W3; break; case 4: W = W4; break; case 5: W = W5; break;
        default: W = W6; break;
    }
    Wt[(size_t)w * (DD * DD) + (size_t)n * DD + k] = (__bf16)W[(size_t)k * DD + n];
}

__global__ void tab_kernel(const int* __restrict__ u, const float* __restrict__ ut,
                           float* __restrict__ tab)
{
    int t = threadIdx.x;           // 512 threads = 64 users * 8 slots
    int b = t >> 3, j = t & 7;
    tab[t] = (j < 5) ? ut[(size_t)u[b] * 5 + j] : 0.0f;
}

// ---------- stage 2: bf16 GEMM  C = A @ W + bias ----------
// One wave computes a 32x64 tile (2 M-tiles x 4 N-tiles = 8 WMMA accumulators),
// so each K-step issues 6 fragment loads for 8 WMMAs (vs 2 loads per WMMA before).
// Block = 4 waves -> 32 rows x 256 cols per block.

__global__ __launch_bounds__(128)
void gemm16_kernel(const __bf16* __restrict__ A, const __bf16* __restrict__ Wt,
                   const float* __restrict__ bias, __bf16* __restrict__ C,
                   float scl, int transposed)
{
    int wid  = threadIdx.x >> 5;       // 0..3
    int lane = threadIdx.x & 31;
    int m0 = blockIdx.x * 32;
    int n0 = wid * 64;

    v8f acc[2][4] = {};
#pragma unroll
    for (int k0 = 0; k0 < DD; k0 += 32) {
        v16bf a0 = load_a(A, DD, m0,      k0);
        v16bf a1 = load_a(A, DD, m0 + 16, k0);
#pragma unroll
        for (int j = 0; j < 4; ++j) {
            v16bf b = load_bt(Wt, DD, n0 + j * 16, k0);
            acc[0][j] = wmma_bf16(a0, b, acc[0][j]);
            acc[1][j] = wmma_bf16(a1, b, acc[1][j]);
        }
    }
    int n = lane & 15, mb = (lane & 16) ? 8 : 0;
#pragma unroll
    for (int i = 0; i < 2; ++i) {
#pragma unroll
        for (int j = 0; j < 4; ++j) {
            float bv = bias[n0 + j * 16 + n];
#pragma unroll
            for (int r = 0; r < 8; ++r) {
                int row = m0 + i * 16 + mb + r;
                float v = (acc[i][j][r] + bv) * scl;
                if (!transposed) {
                    C[(size_t)row * DD + n0 + j * 16 + n] = (__bf16)v;
                } else {               // per-batch transposed store: Vt[b][d][l]
                    int bb = row >> 9, lr = row & (LL - 1);
                    C[(size_t)bb * (DD * LL) + (size_t)(n0 + j * 16 + n) * LL + lr] = (__bf16)v;
                }
            }
        }
    }
}

// ---------- stage 3: fused dual-branch attention with Hamming prior ----------

__global__ __launch_bounds__(512)
void attn_kernel(const __bf16* __restrict__ Qib, const __bf16* __restrict__ Kib,
                 const __bf16* __restrict__ Vtib,
                 const __bf16* __restrict__ Qpb, const __bf16* __restrict__ Kpb,
                 const __bf16* __restrict__ Vtpb,
                 const unsigned char* __restrict__ codes, const float* __restrict__ tab,
                 const float* __restrict__ mask, __bf16* __restrict__ Osum)
{
    __shared__ __align__(16) float  s_sc[16][LL];    // raw scores
    __shared__ __align__(16) float  s_mask[16][LL];  // cached attn_mask tile
    __shared__ __align__(16) float  s_ham[16][LL];   // hamming-prior probs
    __shared__ __align__(16) __bf16 s_p[16][LL];     // probs (bf16) for PV WMMA
    __shared__ unsigned char s_code[LL];
    __shared__ float s_tab[8];

    int b   = blockIdx.y;
    int q0  = blockIdx.x * 16;
    int tid = threadIdx.x;
    int wid = tid >> 5, lane = tid & 31;   // 16 waves; wave r owns query row q0+r

    s_code[tid] = codes[(size_t)b * LL + tid];
    if (tid < 8) s_tab[tid] = tab[b * 8 + tid];
    __syncthreads();

    // hamming prior softmax (shared by both branches) + cache mask tile
    {
        int q = q0 + wid;
        unsigned cq = s_code[q];
        const float* mrow = mask + ((size_t)b * LL + q) * LL;
        float hv[16];
        float mx = -1e30f;
#pragma unroll
        for (int i = 0; i < 16; ++i) {
            int k = lane + 32 * i;
            float mk = mrow[k];
            float h  = s_tab[__popc(cq ^ (unsigned)s_code[k])] * SCALE + mk;
            s_mask[wid][k] = mk;
            hv[i] = h; mx = fmaxf(mx, h);
        }
        mx = wave_max(mx);
        float sum = 0.f;
#pragma unroll
        for (int i = 0; i < 16; ++i) { hv[i] = __expf(hv[i] - mx); sum += hv[i]; }
        float inv = 1.0f / wave_sum(sum);
#pragma unroll
        for (int i = 0; i < 16; ++i) s_ham[wid][lane + 32 * i] = hv[i] * inv;
    }
    __syncthreads();

    v8f oacc = {};                          // O accumulated over BOTH branches
    const __bf16* Qs[2]  = {Qib, Qpb};
    const __bf16* Ks[2]  = {Kib, Kpb};
    const __bf16* Vts[2] = {Vtib, Vtpb};

    for (int br = 0; br < 2; ++br) {
        const __bf16* Q  = Qs[br]  + (size_t)b * LL * DD;   // pre-scaled by 1/sqrt(D)
        const __bf16* K  = Ks[br]  + (size_t)b * LL * DD;
        const __bf16* Vt = Vts[br] + (size_t)b * DD * LL;

        // scores = Q @ K^T : each wave does 2 of the 32 key tiles
        // (A fragments of Q are loop-invariant across t and CSE'd by the compiler)
#pragma unroll
        for (int t = 0; t < 2; ++t) {
            int n0 = (wid + t * 16) * 16;
            v8f acc = {};
#pragma unroll
            for (int k0 = 0; k0 < DD; k0 += 32) {
                v16bf a  = load_a(Q, DD, q0, k0);
                v16bf bb = load_bt(K, DD, n0, k0);   // B(k=d, n=key) = K[key][d]
                acc = wmma_bf16(a, bb, acc);
            }
            int n = lane & 15, mb = (lane & 16) ? 8 : 0;
#pragma unroll
            for (int r = 0; r < 8; ++r) s_sc[mb + r][n0 + n] = acc[r];
        }
        __syncthreads();

        // softmax(scores + mask) + ham_probs -> bf16 probs (wave r = row r)
        {
            float sv[16];
            float mx = -1e30f;
#pragma unroll
            for (int i = 0; i < 16; ++i) {
                int k = lane + 32 * i;
                float s = s_sc[wid][k] + s_mask[wid][k];
                sv[i] = s; mx = fmaxf(mx, s);
            }
            mx = wave_max(mx);
            float sum = 0.f;
#pragma unroll
            for (int i = 0; i < 16; ++i) { sv[i] = __expf(sv[i] - mx); sum += sv[i]; }
            float inv = 1.0f / wave_sum(sum);
#pragma unroll
            for (int i = 0; i < 16; ++i) {
                int k = lane + 32 * i;
                s_p[wid][k] = (__bf16)(sv[i] * inv + s_ham[wid][k]);
            }
        }
        __syncthreads();

        // O += P @ V : wave wid owns d-tile wid (16 waves * 16 = 256 = D)
        {
            int d0 = wid * 16;
#pragma unroll
            for (int k0 = 0; k0 < LL; k0 += 32) {
                v16bf a  = load_a(&s_p[0][0], LL, 0, k0);   // A from LDS
                v16bf bb = load_bt(Vt, LL, d0, k0);         // B(k=key, n=d) = Vt[d][key]
                oacc = wmma_bf16(a, bb, oacc);
            }
        }
        __syncthreads();
    }

    // store O_ib + O_pb (bf16) for the Wf GEMM
    {
        int n = lane & 15, mb = (lane & 16) ? 8 : 0;
        int d0 = wid * 16;
#pragma unroll
        for (int r = 0; r < 8; ++r) {
            int row = b * LL + q0 + mb + r;
            Osum[(size_t)row * DD + d0 + n] = (__bf16)oacc[r];
        }
    }
}

// ---------- stage 4: O @ Wf + bias + residuals, then final LayerNorm ----------

__global__ __launch_bounds__(512)
void final_kernel(const __bf16* __restrict__ Osum, const __bf16* __restrict__ Wft,
                  const float* __restrict__ bf_, const float* __restrict__ seq,
                  const float* __restrict__ bse, const float* __restrict__ lnw,
                  const float* __restrict__ lnb, float* __restrict__ out)
{
    __shared__ __align__(16) float s_x[16][DD];
    int m0  = blockIdx.x * 16;
    int tid = threadIdx.x, wid = tid >> 5, lane = tid & 31;
    int n0  = wid * 16;

    v8f acc = {};
#pragma unroll
    for (int k0 = 0; k0 < DD; k0 += 32) {
        v16bf a = load_a(Osum, DD, m0, k0);
        v16bf b = load_bt(Wft, DD, n0, k0);
        acc = wmma_bf16(a, b, acc);
    }
    int n = lane & 15, mb = (lane & 16) ? 8 : 0;
    float bv = bf_[n0 + n];
#pragma unroll
    for (int r = 0; r < 8; ++r) {
        int row = m0 + mb + r;
        size_t idx = (size_t)row * DD + n0 + n;
        s_x[mb + r][n0 + n] = acc[r] + bv + seq[idx] + bse[idx];
    }
    __syncthreads();

    // LayerNorm: wave wid owns row wid (8 elements per lane)
    float v[8];
    float sum = 0.f;
#pragma unroll
    for (int i = 0; i < 8; ++i) { v[i] = s_x[wid][lane + 32 * i]; sum += v[i]; }
    float mean = wave_sum(sum) * (1.0f / DD);
    float var = 0.f;
#pragma unroll
    for (int i = 0; i < 8; ++i) { v[i] -= mean; var += v[i] * v[i]; }
    float rstd = rsqrtf(wave_sum(var) * (1.0f / DD) + EPS);
#pragma unroll
    for (int i = 0; i < 8; ++i) {
        int c = lane + 32 * i;
        out[(size_t)(m0 + wid) * DD + c] = lnw[c] * v[i] * rstd + lnb[c];
    }
}

// ---------- host launcher ----------

extern "C" void kernel_launch(void* const* d_in, const int* in_sizes, int n_in,
                              void* d_out, int out_size, void* d_ws, size_t ws_size,
                              hipStream_t stream)
{
    const int*   u     = (const int*)d_in[0];
    const float* seq   = (const float*)d_in[1];
    const float* item  = (const float*)d_in[2];
    const float* pos   = (const float*)d_in[3];
    const int*   bsq   = (const int*)d_in[4];
    const float* bse   = (const float*)d_in[5];
    const float* mask  = (const float*)d_in[6];
    const float* utab  = (const float*)d_in[7];
    const float* Wq_ib = (const float*)d_in[8];   const float* bq_ib = (const float*)d_in[9];
    const float* Wk_ib = (const float*)d_in[10];  const float* bk_ib = (const float*)d_in[11];
    const float* Wv_ib = (const float*)d_in[12];  const float* bv_ib = (const float*)d_in[13];
    const float* Wq_pb = (const float*)d_in[14];  const float* bq_pb = (const float*)d_in[15];
    const float* Wk_pb = (const float*)d_in[16];  const float* bk_pb = (const float*)d_in[17];
    const float* Wv_pb = (const float*)d_in[18];  const float* bv_pb = (const float*)d_in[19];
    const float* Wf    = (const float*)d_in[20];  const float* bf_   = (const float*)d_in[21];
    const float* lnw_qib = (const float*)d_in[22]; const float* lnb_qib = (const float*)d_in[23];
    const float* lnw_qpb = (const float*)d_in[24]; const float* lnb_qpb = (const float*)d_in[25];
    const float* lnw_f   = (const float*)d_in[26]; const float* lnb_f   = (const float*)d_in[27];

    char* ws = (char*)d_ws;
    const size_t S = (size_t)NB * LL * DD * sizeof(__bf16);   // 16 MiB per bf16 tensor
    __bf16* Xib  = (__bf16*)(ws + 0 * S);
    __bf16* Xpb  = (__bf16*)(ws + 1 * S);
    __bf16* LNib = (__bf16*)(ws + 2 * S);
    __bf16* LNpb = (__bf16*)(ws + 3 * S);
    __bf16* Qib  = (__bf16*)(ws + 4 * S);
    __bf16* Kib  = (__bf16*)(ws + 5 * S);
    __bf16* Vtib = (__bf16*)(ws + 6 * S);
    __bf16* Qpb  = (__bf16*)(ws + 7 * S);
    __bf16* Kpb  = (__bf16*)(ws + 8 * S);
    __bf16* Vtpb = (__bf16*)(ws + 9 * S);
    __bf16* Os   = (__bf16*)(ws + 10 * S);
    __bf16* Wt   = (__bf16*)(ws + 11 * S);                    // 7 x 256x256 bf16
    unsigned char* codes = (unsigned char*)(ws + 11 * S + 7 * (size_t)DD * DD * 2);
    float* tabp  = (float*)(ws + 11 * S + 7 * (size_t)DD * DD * 2 + (size_t)NB * LL);

    prep_kernel<<<dim3(NB * LL), dim3(256), 0, stream>>>(
        item, pos, bse, bsq, lnw_qib, lnb_qib, lnw_qpb, lnb_qpb,
        Xib, Xpb, LNib, LNpb, codes);

    transpose_w_kernel<<<dim3(DD, 7), dim3(DD), 0, stream>>>(
        Wq_ib, Wk_ib, Wv_ib, Wq_pb, Wk_pb, Wv_pb, Wf, Wt);
    tab_kernel<<<dim3(1), dim3(512), 0, stream>>>(u, utab, tabp);

    dim3 gg(NB * LL / 32), gb(128);
    gemm16_kernel<<<gg, gb, 0, stream>>>(LNib, Wt + 0 * DD * DD, bq_ib, Qib,  SCALE, 0);
    gemm16_kernel<<<gg, gb, 0, stream>>>(Xib,  Wt + 1 * DD * DD, bk_ib, Kib,  1.0f,  0);
    gemm16_kernel<<<gg, gb, 0, stream>>>(Xib,  Wt + 2 * DD * DD, bv_ib, Vtib, 1.0f,  1);
    gemm16_kernel<<<gg, gb, 0, stream>>>(LNpb, Wt + 3 * DD * DD, bq_pb, Qpb,  SCALE, 0);
    gemm16_kernel<<<gg, gb, 0, stream>>>(Xpb,  Wt + 4 * DD * DD, bk_pb, Kpb,  1.0f,  0);
    gemm16_kernel<<<gg, gb, 0, stream>>>(Xpb,  Wt + 5 * DD * DD, bv_pb, Vtpb, 1.0f,  1);

    attn_kernel<<<dim3(LL / 16, NB), dim3(512), 0, stream>>>(
        Qib, Kib, Vtib, Qpb, Kpb, Vtpb, codes, tabp, mask, Os);

    final_kernel<<<dim3(NB * LL / 16), dim3(512), 0, stream>>>(
        Os, Wt + 6 * DD * DD, bf_, seq, bse, lnw_f, lnb_f, (float*)d_out);
}
